// FullMultiHeadAttention_33715493273891
// MI455X (gfx1250) — matrix operand order repaired and checked by reading, not compiled, not run on hardware
//
#include <hip/hip_runtime.h>
#include <hip/hip_bf16.h>

// ---------------------------------------------------------------------------
// FullMultiHeadAttention for MI455X (gfx1250, wave32, WMMA + async-LDS)
//   B=2, Tq=Tk=2048, d_model=1024, H=16, Dh=64
//
// Pipeline:
//   1) cvt: fp32 -> f16 for q,k,v,Wq,Wk,Wv,Wo (biases stay fp32)
//   2) proj_gemm (WMMA f16, 2x-unrolled dual-buffer pipeline):
//        Q,K -> [B,H,T,64], V -> [B,H,64,T]
//   3) flash_attn (WMMA f16; K/V tiles double-buffered in LDS via
//      global_load_async_to_lds_b128; bank-conflict-free padded tiles;
//      streaming softmax; scores never hit HBM)
//   4) out_proj (WMMA f16 -> fp32 output)
// ---------------------------------------------------------------------------

typedef __attribute__((ext_vector_type(16))) _Float16 v16h;
typedef __attribute__((ext_vector_type(8)))  _Float16 v8h;
typedef __attribute__((ext_vector_type(4)))  _Float16 v4h;
typedef __attribute__((ext_vector_type(8)))  float    v8f;
typedef __attribute__((ext_vector_type(4)))  float    v4f;

#define D_MODEL 1024
#define N_HEAD  16
#define D_HEAD  64
#define BATCH   2
#define SEQ     2048
#define M_ROWS  (BATCH * SEQ)   // 4096 rows for all projection GEMMs

#define KPAD 72   // padded K-tile row stride (halves): 144B = 36 banks
#define VPAD 40   // padded V-tile row stride (halves):  80B = 20 banks

__device__ __forceinline__ v8f wmma_f16(v16h a, v16h b, v8f c) {
  // v_wmma_f32_16x16x32_f16  (neg_a, A, neg_b, B, c_mod, C, reuse_a, reuse_b)
  return __builtin_amdgcn_wmma_f32_16x16x32_f16(false, a, false, b, (short)0, c,
                                                false, false);
}

// Two 16B loads -> one 16-half fragment (elements 0..7 from lo, 8..15 from hi).
__device__ __forceinline__ v16h pack16(const _Float16* lo, const _Float16* hi) {
  v8h a = *(const v8h*)lo;
  v8h b = *(const v8h*)hi;
  v16h r;
#pragma unroll
  for (int i = 0; i < 8; ++i) { r[i] = a[i]; r[i + 8] = b[i]; }
  return r;
}

// Generic pointers to LDS are {shared_aperture, offset}; low 32 bits are the
// LDS byte address the async-DMA dst operand wants.
__device__ __forceinline__ unsigned lds_addr(const void* p) {
  return (unsigned)(uintptr_t)p;
}

// ASYNCcnt-tracked 16B global->LDS copy (no VGPR round trip).
__device__ __forceinline__ void async_copy_b128(unsigned lds_off, const void* g) {
  asm volatile("global_load_async_to_lds_b128 %0, %1, off"
               :: "v"(lds_off), "v"(g) : "memory");
}

// ---------------------------------------------------------------------------
// 1) fp32 -> f16 conversion (vectorized x4)
// ---------------------------------------------------------------------------
__global__ void cvt_f32_to_f16(const float* __restrict__ src,
                               _Float16* __restrict__ dst, int n4) {
  int i = blockIdx.x * blockDim.x + threadIdx.x;
  if (i < n4) {
    v4f x = ((const v4f*)src)[i];
    v4h y;
#pragma unroll
    for (int j = 0; j < 4; ++j) y[j] = (_Float16)x[j];
    ((v4h*)dst)[i] = y;
  }
}

// ---------------------------------------------------------------------------
// 2) Projection GEMM: out = X @ W^T + b  (X:[4096,1024] f16, W:[1024,1024] f16)
//    Wave computes 16(M) x 64(N). k-loop unrolled 2x with two named buffer
//    sets so loads land directly in their registers: no rotation moves, no
//    WMMA->VALU hazard NOPs, and ~10 loads in flight over each WMMA group.
//    mode 0: scatter to [B,H,T,64] (Q,K); mode 1: scatter to [B,H,64,T] (V^T)
// ---------------------------------------------------------------------------
__global__ void proj_gemm(const _Float16* __restrict__ X,
                          const _Float16* __restrict__ W,
                          const float* __restrict__ bias,
                          _Float16* __restrict__ out, int mode) {
  const int lane = threadIdx.x & 31;
  const int wave = threadIdx.x >> 5;
  const int m0 = blockIdx.x * 16;
  const int n0 = blockIdx.y * 256 + wave * 64;

  const int arow = m0 + (lane & 15);       // A: row per lane
  const int koff = (lane >> 4) * 16;       // A: K sub-offset per half-wave
  const int ncol = lane & 15;              // B/C/D: column per lane
  const int bko  = (lane >> 4) * 16;       // B: K sub-offset per half-wave

  v8f acc[4] = {v8f{}, v8f{}, v8f{}, v8f{}};
  const _Float16* xrow = X + (size_t)arow * D_MODEL;
  const _Float16* wrow[4];
#pragma unroll
  for (int j = 0; j < 4; ++j)
    wrow[j] = W + (size_t)(n0 + j * 16 + ncol) * D_MODEL + bko;

  v16h a0, a1, b0[4], b1[4];
  a0 = pack16(xrow + koff, xrow + 8 + koff);
#pragma unroll
  for (int j = 0; j < 4; ++j) b0[j] = *(const v16h*)(wrow[j]);

  for (int k = 0; k < D_MODEL; k += 64) {
    // stage k+32 into buffer set 1 (always in range: k <= 960 -> k+32 <= 992)
    __builtin_prefetch(xrow + k + 288, 0, 1);           // global_prefetch_b8
    a1 = pack16(xrow + k + 32 + koff, xrow + k + 40 + koff);
#pragma unroll
    for (int j = 0; j < 4; ++j) b1[j] = *(const v16h*)(wrow[j] + k + 32);

#pragma unroll
    for (int j = 0; j < 4; ++j) acc[j] = wmma_f16(a0, b0[j], acc[j]);

    // stage k+64 into buffer set 0
    if (k + 64 < D_MODEL) {
      a0 = pack16(xrow + k + 64 + koff, xrow + k + 72 + koff);
#pragma unroll
      for (int j = 0; j < 4; ++j) b0[j] = *(const v16h*)(wrow[j] + k + 64);
    }

#pragma unroll
    for (int j = 0; j < 4; ++j) acc[j] = wmma_f16(a1, b1[j], acc[j]);
  }

  // Epilogue: C layout = VGPR i holds row m0 + (lane>>4)*8 + i, col n0+j*16+ncol
  const int rbase = (lane >> 4) * 8;
#pragma unroll
  for (int j = 0; j < 4; ++j) {
    const int n = n0 + j * 16 + ncol;
    const float bv = bias[n];
    const int h = n >> 6, d = n & 63;
#pragma unroll
    for (int i = 0; i < 8; ++i) {
      const int m = m0 + rbase + i;
      const int bb = m >> 11, t = m & (SEQ - 1);
      const float val = acc[j][i] + bv;
      size_t idx;
      if (mode == 0) idx = ((((size_t)bb * N_HEAD + h) * SEQ) + t) * D_HEAD + d;
      else           idx = ((((size_t)bb * N_HEAD + h) * D_HEAD) + d) * SEQ + t;
      out[idx] = (_Float16)val;
    }
  }
}

// ---------------------------------------------------------------------------
// 3) Flash attention. Grid: (SEQ/64, B*H), block 128 (4 waves, 16 q-rows each).
//    K (32x64) and V^T (64x32) tiles are double-buffered in padded LDS and
//    filled by ASYNCcnt-tracked global_load_async_to_lds_b128.
// ---------------------------------------------------------------------------
__global__ void flash_attn(const _Float16* __restrict__ Qh,
                           const _Float16* __restrict__ Kh,
                           const _Float16* __restrict__ Vt,
                           const int* __restrict__ mask,
                           _Float16* __restrict__ ctx) {
  __shared__ __align__(16) _Float16 lds_k[2][32][KPAD];  // 9 KB
  __shared__ __align__(16) _Float16 lds_v[2][64][VPAD];  // 10 KB
  __shared__ __align__(16) _Float16 lds_p[4][16][40];    // per-wave P reshape

  const int tid  = threadIdx.x;
  const int lane = tid & 31;
  const int wave = tid >> 5;
  const int bh = blockIdx.y;              // b*16 + h
  const int b = bh >> 4;
  const int h = bh & 15;
  const int q0 = blockIdx.x * 64 + wave * 16;

  const _Float16* Qp = Qh + ((size_t)bh * SEQ + q0) * D_HEAD;
  const _Float16* Kp = Kh + (size_t)bh * SEQ * D_HEAD;
  const _Float16* Vp = Vt + (size_t)bh * D_HEAD * SEQ;
  const int* mp = mask + ((size_t)b * SEQ + q0) * SEQ;

  const int arow = lane & 15;
  const int koff = (lane >> 4) * 16;
  const int ncol = lane & 15;
  const int bko  = (lane >> 4) * 16;
  const int rbase = (lane >> 4) * 8;

  // Q A-fragments for Dh=64 (two K=32 slices), resident for the whole loop.
  const _Float16* qrow = Qp + arow * D_HEAD;
  const v16h aq0 = pack16(qrow + 0  + koff, qrow + 8 + koff);
  const v16h aq1 = pack16(qrow + 32 + koff, qrow + 40 + koff);

  v8f acc[4] = {v8f{}, v8f{}, v8f{}, v8f{}};
  float mrow[8], lrow[8];
#pragma unroll
  for (int i = 0; i < 8; ++i) { mrow[i] = -1e30f; lrow[i] = 0.0f; }

  _Float16* lp = &lds_p[wave][0][0];
  const float sc = 0.125f;  // 1/sqrt(64)

  // Cooperative async stage of one 32-wide K/V tile into buffer bb.
  auto stage = [&](int k0, int bb) {
    // K tile: 32 rows x 64 halves = 256 16B chunks; 2 per thread.
#pragma unroll
    for (int c = 0; c < 2; ++c) {
      const int ch = tid + c * 128;
      const int r = ch >> 3, o = (ch & 7) * 8;
      async_copy_b128(lds_addr(&lds_k[bb][r][o]),
                      Kp + (size_t)(k0 + r) * D_HEAD + o);
    }
    // V^T tile: 64 rows x 32 halves = 256 16B chunks; 2 per thread.
#pragma unroll
    for (int c = 0; c < 2; ++c) {
      const int ch = tid + c * 128;
      const int r = ch >> 2, o = (ch & 3) * 8;
      async_copy_b128(lds_addr(&lds_v[bb][r][o]),
                      Vp + (size_t)r * SEQ + k0 + o);
    }
  };

  stage(0, 0);
  for (int kt = 0; kt < SEQ / 32; ++kt) {
    const int k0 = kt * 32;
    asm volatile("s_wait_asynccnt 0" ::: "memory");  // this wave's DMA done
    __syncthreads();                                 // whole tile visible
    if (kt + 1 < SEQ / 32) stage(k0 + 32, (kt + 1) & 1);

    const _Float16* lk = &lds_k[kt & 1][0][0];
    const _Float16* lv = &lds_v[kt & 1][0][0];

    // ---- S = Q @ K^T for 32 k-columns (two 16x16 tiles), K from LDS ----
    v8f s0 = {}, s1 = {};
    const _Float16* kr0 = lk + (size_t)ncol * KPAD + bko;
    const _Float16* kr1 = kr0 + 16 * KPAD;
    s0 = wmma_f16(aq0, pack16(kr0, kr0 + 8), s0);
    s0 = wmma_f16(aq1, pack16(kr0 + 32, kr0 + 40), s0);
    s1 = wmma_f16(aq0, pack16(kr1, kr1 + 8), s1);
    s1 = wmma_f16(aq1, pack16(kr1 + 32, kr1 + 40), s1);

    // ---- scale + mask + per-row tile max (rows live across 16 lanes) ----
    float tmax[8];
#pragma unroll
    for (int i = 0; i < 8; ++i) {
      const int* mr = mp + (size_t)(rbase + i) * SEQ + k0;
      float x0 = s0[i] * sc; if (mr[ncol] == 0)      x0 = -1e30f;
      float x1 = s1[i] * sc; if (mr[16 + ncol] == 0) x1 = -1e30f;
      s0[i] = x0; s1[i] = x1;
      float t = fmaxf(x0, x1);
#pragma unroll
      for (int m = 1; m < 16; m <<= 1) t = fmaxf(t, __shfl_xor(t, m, 32));
      tmax[i] = t;
    }

    // ---- online softmax update ----
    float corr[8];
#pragma unroll
    for (int i = 0; i < 8; ++i) {
      const float mnew = fmaxf(mrow[i], tmax[i]);
      corr[i] = __expf(mrow[i] - mnew);
      const float p0 = __expf(s0[i] - mnew);
      const float p1 = __expf(s1[i] - mnew);
      float rs = p0 + p1;
#pragma unroll
      for (int m = 1; m < 16; m <<= 1) rs += __shfl_xor(rs, m, 32);
      lrow[i] = lrow[i] * corr[i] + rs;
      mrow[i] = mnew;
      s0[i] = p0; s1[i] = p1;
    }
#pragma unroll
    for (int j = 0; j < 4; ++j)
#pragma unroll
      for (int i = 0; i < 8; ++i) acc[j][i] *= corr[i];

    // ---- reshape P: C layout -> A layout via per-wave LDS tile ----
#pragma unroll
    for (int i = 0; i < 8; ++i) {
      lp[(rbase + i) * 40 + ncol]      = (_Float16)s0[i];
      lp[(rbase + i) * 40 + 16 + ncol] = (_Float16)s1[i];
    }
    asm volatile("s_wait_dscnt 0" ::: "memory");  // wave-internal LDS RAW
    v16h pa = pack16(lp + arow * 40 + koff, lp + arow * 40 + 8 + koff);

    // ---- O += P @ V  (V^T tile from LDS: contiguous per-lane B loads) ----
#pragma unroll
    for (int j = 0; j < 4; ++j) {
      const _Float16* vp = lv + (size_t)(j * 16 + ncol) * VPAD + bko;
      acc[j] = wmma_f16(pa, pack16(vp, vp + 8), acc[j]);
    }
    __syncthreads();   // all waves done reading before buffer is re-staged
  }

  // ---- normalize and store f16 context in [B, Tq, d_model] layout ----
#pragma unroll
  for (int j = 0; j < 4; ++j) {
#pragma unroll
    for (int i = 0; i < 8; ++i) {
      const float o = acc[j][i] / lrow[i];
      ctx[((size_t)b * SEQ + q0 + rbase + i) * D_MODEL +
          h * D_HEAD + j * 16 + ncol] = (_Float16)o;
    }
  }
}

// ---------------------------------------------------------------------------
// 4) Output projection: out = ctx @ Wo^T + bo  (fp32 output, same pipeline)
// ---------------------------------------------------------------------------
__global__ void out_proj(const _Float16* __restrict__ X,
                         const _Float16* __restrict__ W,
                         const float* __restrict__ bias,
                         float* __restrict__ out) {
  const int lane = threadIdx.x & 31;
  const int wave = threadIdx.x >> 5;
  const int m0 = blockIdx.x * 16;
  const int n0 = blockIdx.y * 256 + wave * 64;

  const int koff = (lane >> 4) * 16;
  const int ncol = lane & 15;
  const int bko  = (lane >> 4) * 16;

  v8f acc[4] = {v8f{}, v8f{}, v8f{}, v8f{}};
  const _Float16* xrow = X + (size_t)(m0 + (lane & 15)) * D_MODEL;
  const _Float16* wrow[4];
#pragma unroll
  for (int j = 0; j < 4; ++j)
    wrow[j] = W + (size_t)(n0 + j * 16 + ncol) * D_MODEL + bko;

  v16h a0, a1, b0[4], b1[4];
  a0 = pack16(xrow + koff, xrow + 8 + koff);
#pragma unroll
  for (int j = 0; j < 4; ++j) b0[j] = *(const v16h*)(wrow[j]);

  for (int k = 0; k < D_MODEL; k += 64) {
    __builtin_prefetch(xrow + k + 288, 0, 1);
    a1 = pack16(xrow + k + 32 + koff, xrow + k + 40 + koff);
#pragma unroll
    for (int j = 0; j < 4; ++j) b1[j] = *(const v16h*)(wrow[j] + k + 32);

#pragma unroll
    for (int j = 0; j < 4; ++j) acc[j] = wmma_f16(a0, b0[j], acc[j]);

    if (k + 64 < D_MODEL) {
      a0 = pack16(xrow + k + 64 + koff, xrow + k + 72 + koff);
#pragma unroll
      for (int j = 0; j < 4; ++j) b0[j] = *(const v16h*)(wrow[j] + k + 64);
    }

#pragma unroll
    for (int j = 0; j < 4; ++j) acc[j] = wmma_f16(a1, b1[j], acc[j]);
  }

  const int rbase = (lane >> 4) * 8;
#pragma unroll
  for (int j = 0; j < 4; ++j) {
    const int n = n0 + j * 16 + ncol;
    const float bv = bias[n];
#pragma unroll
    for (int i = 0; i < 8; ++i)
      out[(size_t)(m0 + rbase + i) * D_MODEL + n] = acc[j][i] + bv;
  }
}

// ---------------------------------------------------------------------------
extern "C" void kernel_launch(void* const* d_in, const int* in_sizes, int n_in,
                              void* d_out, int out_size, void* d_ws, size_t ws_size,
                              hipStream_t stream) {
  const float* q  = (const float*)d_in[0];
  const float* k  = (const float*)d_in[1];
  const float* v  = (const float*)d_in[2];
  const int*   mask = (const int*)d_in[3];
  const float* Wq = (const float*)d_in[4];
  const float* bq = (const float*)d_in[5];
  const float* Wk = (const float*)d_in[6];
  const float* bk = (const float*)d_in[7];
  const float* Wv = (const float*)d_in[8];
  const float* bv = (const float*)d_in[9];
  const float* Wo = (const float*)d_in[10];
  const float* bo = (const float*)d_in[11];

  // Workspace layout (f16 elements): 3 activations + 4 weights + Q/K/Vt/ctx
  const size_t E  = (size_t)M_ROWS * D_MODEL;    // 4194304
  const size_t Wn = (size_t)D_MODEL * D_MODEL;   // 1048576
  _Float16* ws  = (_Float16*)d_ws;
  _Float16* hq  = ws;
  _Float16* hk  = hq + E;
  _Float16* hv  = hk + E;
  _Float16* wq  = hv + E;
  _Float16* wk  = wq + Wn;
  _Float16* wv  = wk + Wn;
  _Float16* wo  = wv + Wn;
  _Float16* Qh  = wo + Wn;
  _Float16* Kh  = Qh + E;
  _Float16* Vt  = Kh + E;
  _Float16* ctx = Vt + E;                         // total 64 MB

  // 1) conversions
  {
    const int n4a = (int)(E / 4), n4w = (int)(Wn / 4);
    dim3 blk(256);
    dim3 ga((n4a + 255) / 256), gw((n4w + 255) / 256);
    cvt_f32_to_f16<<<ga, blk, 0, stream>>>(q, hq, n4a);
    cvt_f32_to_f16<<<ga, blk, 0, stream>>>(k, hk, n4a);
    cvt_f32_to_f16<<<ga, blk, 0, stream>>>(v, hv, n4a);
    cvt_f32_to_f16<<<gw, blk, 0, stream>>>(Wq, wq, n4w);
    cvt_f32_to_f16<<<gw, blk, 0, stream>>>(Wk, wk, n4w);
    cvt_f32_to_f16<<<gw, blk, 0, stream>>>(Wv, wv, n4w);
    cvt_f32_to_f16<<<gw, blk, 0, stream>>>(Wo, wo, n4w);
  }

  // 2) projections (Q/K row-major per head, V transposed per head)
  {
    dim3 grid(M_ROWS / 16, D_MODEL / 256), blk(128);
    proj_gemm<<<grid, blk, 0, stream>>>(hq, wq, bq, Qh, 0);
    proj_gemm<<<grid, blk, 0, stream>>>(hk, wk, bk, Kh, 0);
    proj_gemm<<<grid, blk, 0, stream>>>(hv, wv, bv, Vt, 1);
  }

  // 3) flash attention (async-LDS double-buffered K/V tiles)
  flash_attn<<<dim3(SEQ / 64, BATCH * N_HEAD), 128, 0, stream>>>(Qh, Kh, Vt, mask, ctx);

  // 4) output projection (fp32 result)
  out_proj<<<dim3(M_ROWS / 16, D_MODEL / 256), 128, 0, stream>>>(ctx, wo, bo,
                                                                 (float*)d_out);
}
